// PIDSparseAttention_5531917877775
// MI455X (gfx1250) — matrix operand order repaired and checked by reading, not compiled, not run on hardware
//
#include <hip/hip_runtime.h>
#include <hip/hip_bf16.h>
#include <math.h>

// ---------------------------------------------------------------------------
// Problem constants (match reference)
// ---------------------------------------------------------------------------
#define BB   2
#define TT   4096
#define DD   1024
#define HH   16
#define DHH  64
#define AA   16           // dilated positions per query
#define MM   (BB * TT)    // 8192 token rows
#define MAX_GATE 0.6f

typedef _Float16 half8  __attribute__((ext_vector_type(8)));
typedef _Float16 v16h   __attribute__((ext_vector_type(16)));
typedef float    v8f    __attribute__((ext_vector_type(8)));
typedef int      b128_t __attribute__((vector_size(16)));   // builtin's pointee

// ---- CDNA5 async global->LDS staging (probe via __has_builtin) ------------
#ifndef __has_builtin
#define __has_builtin(x) 0
#endif
#if __has_builtin(__builtin_amdgcn_global_load_async_to_lds_b128)
#define HAS_ASYNC 1
#else
#define HAS_ASYNC 0
#endif

#if HAS_ASYNC
#if __has_builtin(__builtin_amdgcn_s_wait_asynccnt)
#define WAIT_ASYNC(n) __builtin_amdgcn_s_wait_asynccnt(n)
#else
#define WAIT_ASYNC(n) asm volatile("s_wait_asynccnt %0" ::"i"(n))
#endif
#endif

// ---------------------------------------------------------------------------
// Kernel 1: PID scan — integral (running mean) and derivative (first diff),
// emitted directly as the stacked f16 GEMM operand Acat[3][M][D]:
//   s=0: x, s=1: integral, s=2: derivative.
// One thread per (b,d) column; consecutive threads -> consecutive d.
// ---------------------------------------------------------------------------
__global__ __launch_bounds__(256) void pid_scan(const float* __restrict__ x,
                                                _Float16* __restrict__ Acat) {
  int col = blockIdx.x * blockDim.x + threadIdx.x;   // over B*D
  if (col >= BB * DD) return;
  int b = col / DD;
  int d = col - b * DD;
  const size_t MD = (size_t)MM * DD;
  float run = 0.0f, prev = 0.0f;
  for (int t = 0; t < TT; ++t) {
    size_t row = (size_t)b * TT + t;
    size_t idx = row * DD + d;
    float v = x[idx];
    run += v;
    Acat[idx]          = (_Float16)v;
    Acat[idx + MD]     = (_Float16)(run / (float)(t + 1));
    Acat[idx + 2 * MD] = (_Float16)((t == 0) ? 0.0f : (v - prev));
    prev = v;
  }
}

// ---------------------------------------------------------------------------
// Kernel 2: PID gates — per token: softmax(x@Wg + bg), clamp 0.6, renorm.
// One wave32 per token; shuffle tree reduction over D=1024.
// ---------------------------------------------------------------------------
__global__ __launch_bounds__(256) void pid_gates(const float* __restrict__ x,
                                                 const float* __restrict__ Wg,
                                                 const float* __restrict__ bg,
                                                 float* __restrict__ g) {
  int grp  = (blockIdx.x * blockDim.x + threadIdx.x) >> 5;   // token row
  int lane = threadIdx.x & 31;
  if (grp >= MM) return;
  const float* xr = x + (size_t)grp * DD;
  float s0 = 0.f, s1 = 0.f, s2 = 0.f;
  for (int d = lane; d < DD; d += 32) {
    float xv = xr[d];
    s0 += xv * Wg[d * 3 + 0];
    s1 += xv * Wg[d * 3 + 1];
    s2 += xv * Wg[d * 3 + 2];
  }
  for (int off = 16; off >= 1; off >>= 1) {
    s0 += __shfl_xor(s0, off, 32);
    s1 += __shfl_xor(s1, off, 32);
    s2 += __shfl_xor(s2, off, 32);
  }
  float l0 = s0 + bg[0], l1 = s1 + bg[1], l2 = s2 + bg[2];
  float mx = fmaxf(l0, fmaxf(l1, l2));
  float e0 = expf(l0 - mx), e1 = expf(l1 - mx), e2 = expf(l2 - mx);
  float es = e0 + e1 + e2;
  float g0 = fminf(e0 / es, MAX_GATE);
  float g1 = fminf(e1 / es, MAX_GATE);
  float g2 = fminf(e2 / es, MAX_GATE);
  float gs = g0 + g1 + g2;
  if (lane == 0) {
    g[(size_t)grp * 3 + 0] = g0 / gs;
    g[(size_t)grp * 3 + 1] = g1 / gs;
    g[(size_t)grp * 3 + 2] = g2 / gs;
  }
}

// ---------------------------------------------------------------------------
// Kernel 3: weight transpose + f32->f16 convert: W[K][N] -> Wt[N][K].
// Pre-transposing makes the GEMM B-fragment a contiguous ds_load_b128 pair.
// ---------------------------------------------------------------------------
__global__ __launch_bounds__(256) void wt_convert(const float* __restrict__ W,
                                                  _Float16* __restrict__ Wt,
                                                  int K, int N) {
  __shared__ float tile[32][33];
  int k0 = blockIdx.y * 32, n0 = blockIdx.x * 32;
  for (int i = threadIdx.y; i < 32; i += 8)
    tile[i][threadIdx.x] = W[(size_t)(k0 + i) * N + n0 + threadIdx.x];
  __syncthreads();
  for (int i = threadIdx.y; i < 32; i += 8)
    Wt[(size_t)(n0 + i) * K + k0 + threadIdx.x] = (_Float16)tile[threadIdx.x][i];
}

// ---------------------------------------------------------------------------
// Kernel 4: multi-source gated WMMA GEMM (all-f16 operands, f32 accum/out).
//   C[M,N] = sum_s gate_s(row) * (A[s][M][K] @ Wt[s][N][K]^T)
// Block tile 128x128, BK=32 (== WMMA K), 8 waves (2x4), wave = 64x32 = 4x2
// v_wmma_f32_16x16x32_f16 accumulators. Tiles staged into double-buffered
// LDS via CDNA5 async global->LDS b128 DMA (ASYNCcnt pipelined); per-row
// gates folded into the A fragment with packed f16 multiplies (co-executes
// with XDL WMMA). Fragment lane layouts per CDNA5 ISA 7.12.2 (wave32).
// ---------------------------------------------------------------------------
#define BM 128
#define BN 128
#define BK 32

template <int S, bool G>
__global__ __launch_bounds__(256) void gemm_f16(
    const _Float16* __restrict__ A,    // [S*M][K] stacked
    const _Float16* __restrict__ Wt,   // [S][N][K] stacked, pre-transposed
    const float* __restrict__ gates,   // [M][3] (used when G)
    float* __restrict__ C, int M, int N, int K) {
  __shared__ __align__(64) _Float16 lA[2][BM][BK];   // 2 x 8 KB
  __shared__ __align__(64) _Float16 lB[2][BN][BK];   // 2 x 8 KB

  const int tid   = threadIdx.x;
  const int lane  = tid & 31;
  const int wave  = tid >> 5;      // 0..7
  const int waveM = wave >> 2;     // 0..1 -> 64-row slab
  const int waveN = wave & 3;      // 0..3 -> 32-col slab
  const int bm = blockIdx.y * BM;
  const int bn = blockIdx.x * BN;
  const int hi  = lane >> 4;       // half-wave select
  const int l15 = lane & 15;

  const int kpc    = K / BK;       // k-chunks per source
  const int CHUNKS = S * kpc;

  v8f acc[4][2];
#pragma unroll
  for (int i = 0; i < 4; ++i)
#pragma unroll
    for (int j = 0; j < 2; ++j)
      acc[i][j] = (v8f){0.f, 0.f, 0.f, 0.f, 0.f, 0.f, 0.f, 0.f};

  // stage one k-chunk (A + B tiles, 4 x b128 per thread) into buffer `buf`
  auto stage = [&](int c, int buf) {
    int s  = c / kpc;
    int kk = (c - s * kpc) * BK;
    const _Float16* Ab = A + (size_t)s * M * K;
    const _Float16* Bb = Wt + (size_t)s * N * K;
#pragma unroll
    for (int it = 0; it < 2; ++it) {
      int slot = tid * 2 + it;        // 0..511 : 128 rows x 4 16B segments
      int row  = slot >> 2;
      int seg  = slot & 3;
      const _Float16* ga = Ab + (size_t)(bm + row) * K + kk + seg * 8;
      const _Float16* gb = Bb + (size_t)(bn + row) * K + kk + seg * 8;
      _Float16* la = &lA[buf][row][seg * 8];
      _Float16* lb = &lB[buf][row][seg * 8];
#if HAS_ASYNC
      __builtin_amdgcn_global_load_async_to_lds_b128((b128_t*)ga, (b128_t*)la,
                                                     0, 0);
      __builtin_amdgcn_global_load_async_to_lds_b128((b128_t*)gb, (b128_t*)lb,
                                                     0, 0);
#else
      *(half8*)la = *(const half8*)ga;
      *(half8*)lb = *(const half8*)gb;
#endif
    }
  };

  _Float16 ghc[4];
  int scur = -1;

  // consume buffer `buf` for chunk `c`: 6 fragment loads + 8 WMMAs
  auto compute = [&](int c, int buf) {
    if (G) {
      int s = c / kpc;
      if (s != scur) {
        scur = s;
#pragma unroll
        for (int mt = 0; mt < 4; ++mt) {
          int grow = bm + waveM * 64 + mt * 16 + l15;
          ghc[mt] = (_Float16)gates[(size_t)grow * 3 + s];
        }
      }
    }
    v16h aF[4], bF[2];
#pragma unroll
    for (int mt = 0; mt < 4; ++mt) {
      int row = waveM * 64 + mt * 16 + l15;   // A lane row = M
      // A 16x32 f16: e0..7 -> K = 8*hi + e ; e8..15 -> K = 16 + 8*hi + (e-8)
      union { v16h v; half8 h[2]; } u;
      u.h[0] = *(const half8*)(&lA[buf][row][hi * 8]);
      u.h[1] = *(const half8*)(&lA[buf][row][16 + hi * 8]);
      if (G) {                                // fold per-row gate into A
        u.h[0] = u.h[0] * ghc[mt];
        u.h[1] = u.h[1] * ghc[mt];
      }
      aF[mt] = u.v;
    }
#pragma unroll
    for (int nt = 0; nt < 2; ++nt) {
      int col = waveN * 32 + nt * 16 + l15;   // B lane col = N
      // B 32x16 f16: e0..7 -> K = 8*hi + e ; e8..15 -> K = 16 + 8*hi + (e-8)
      union { v16h v; half8 h[2]; } ub;
      ub.h[0] = *(const half8*)(&lB[buf][col][hi * 8]);
      ub.h[1] = *(const half8*)(&lB[buf][col][16 + hi * 8]);
      bF[nt] = ub.v;
    }
#pragma unroll
    for (int mt = 0; mt < 4; ++mt)
#pragma unroll
      for (int nt = 0; nt < 2; ++nt)
        acc[mt][nt] = __builtin_amdgcn_wmma_f32_16x16x32_f16(
            false, aF[mt], false, bF[nt], (short)0, acc[mt][nt], false, false);
  };

#if HAS_ASYNC
  // double-buffered ASYNCcnt pipeline: stage(c+1) overlaps compute(c)
  stage(0, 0);
  for (int c = 0; c < CHUNKS; ++c) {
    int buf = c & 1;
    if (c + 1 < CHUNKS) {
      stage(c + 1, buf ^ 1);
      WAIT_ASYNC(4);               // 4 newest outstanding = next stage
    } else {
      WAIT_ASYNC(0);
    }
    __syncthreads();               // stage(c) visible to all waves
    compute(c, buf);
    __syncthreads();               // all waves done reading buf before reuse
  }
#else
  for (int c = 0; c < CHUNKS; ++c) {
    __syncthreads();
    stage(c, 0);
    __syncthreads();
    compute(c, 0);
  }
#endif

  // epilogue: C/D layout VGPR r -> M = r + 8*hi, N = lane&15
#pragma unroll
  for (int mt = 0; mt < 4; ++mt) {
    int rowb = bm + waveM * 64 + mt * 16 + hi * 8;
#pragma unroll
    for (int nt = 0; nt < 2; ++nt) {
      int col = bn + waveN * 32 + nt * 16 + l15;
#pragma unroll
      for (int r = 0; r < 8; ++r)
        C[(size_t)(rowb + r) * N + col] = acc[mt][nt][r];
    }
  }
}

// ---------------------------------------------------------------------------
// Kernel 5: dilated sparse attention. One wave32 per (b, h, t).
// Lanes pair up: position a = lane>>1, each lane dots 32 of the 64 dims.
// Softmax over A=16 positions via shuffle tree; output written as f16 rows
// (the direct operand format of the final WMMA projection).
// ---------------------------------------------------------------------------
__global__ __launch_bounds__(256) void sparse_attn(
    const float* __restrict__ q, const float* __restrict__ k,
    const float* __restrict__ v, const int* __restrict__ pos,
    const unsigned char* __restrict__ msk,   // jnp bool_ -> 1 byte
    _Float16* __restrict__ out16) {
  int wid  = (blockIdx.x * blockDim.x + threadIdx.x) >> 5;
  int lane = threadIdx.x & 31;
  if (wid >= BB * HH * TT) return;
  int t  = wid % TT;
  int bh = wid / TT;
  int h  = bh % HH;
  int b  = bh / HH;

  int a   = lane >> 1;
  int sub = lane & 1;
  size_t qbase = ((size_t)(b * TT + t)) * DD + h * DHH;
  int p    = pos[t * AA + a];
  int mval = (int)msk[t * AA + a];
  size_t kbase = ((size_t)(b * TT + p)) * DD + h * DHH;

  float s = 0.f;
  for (int j = 0; j < 32; ++j) {
    int d = sub * 32 + j;
    s += q[qbase + d] * k[kbase + d];
  }
  s += __shfl_xor(s, 1, 32);                 // full 64-dim dot, dup per pair

  float score = mval ? s * 0.125f : -3.4e38f;   // scale = 1/sqrt(64)
  float mx = score;
  for (int off = 2; off <= 16; off <<= 1)    // reduce across 16 pair-groups
    mx = fmaxf(mx, __shfl_xor(mx, off, 32));
  float e = mval ? expf(score - mx) : 0.f;
  float es = e;
  for (int off = 2; off <= 16; off <<= 1)
    es += __shfl_xor(es, off, 32);
  float w = e / es;

  float o0 = 0.f, o1 = 0.f;
  for (int a2 = 0; a2 < AA; ++a2) {
    float wa = __shfl(w, a2 * 2, 32);
    int   pa = __shfl(p, a2 * 2, 32);
    size_t vb = ((size_t)(b * TT + pa)) * DD + h * DHH;
    o0 += wa * v[vb + lane];
    o1 += wa * v[vb + 32 + lane];
  }
  out16[qbase + lane]      = (_Float16)o0;
  out16[qbase + 32 + lane] = (_Float16)o1;
}

// ---------------------------------------------------------------------------
// Host-side launcher
// ---------------------------------------------------------------------------
extern "C" void kernel_launch(void* const* d_in, const int* in_sizes, int n_in,
                              void* d_out, int out_size, void* d_ws,
                              size_t ws_size, hipStream_t stream) {
  const float* x    = (const float*)d_in[0];
  const float* Wp_q = (const float*)d_in[1];
  const float* Wi_q = (const float*)d_in[2];
  const float* Wd_q = (const float*)d_in[3];
  const float* Wg_q = (const float*)d_in[4];
  const float* bg_q = (const float*)d_in[5];
  const float* Wp_k = (const float*)d_in[6];
  const float* Wi_k = (const float*)d_in[7];
  const float* Wd_k = (const float*)d_in[8];
  const float* Wg_k = (const float*)d_in[9];
  const float* bg_k = (const float*)d_in[10];
  const float* Wp_v = (const float*)d_in[11];
  const float* Wi_v = (const float*)d_in[12];
  const float* Wd_v = (const float*)d_in[13];
  const float* Wg_v = (const float*)d_in[14];
  const float* bg_v = (const float*)d_in[15];
  const float* Wo   = (const float*)d_in[16];
  const int* positions      = (const int*)d_in[17];
  const unsigned char* mask = (const unsigned char*)d_in[18];
  float* outp = (float*)d_out;

  // ---- workspace layout ----
  const size_t MD = (size_t)MM * DD;          // 8M elements
  const size_t WD = (size_t)DD * DD;          // 1M elements
  char* w = (char*)d_ws;
  _Float16* Acat = (_Float16*)w;  w += 3 * MD * sizeof(_Float16);  // 48 MB
  _Float16* Wtq  = (_Float16*)w;  w += 3 * WD * sizeof(_Float16);  //  6 MB
  _Float16* Wtk  = (_Float16*)w;  w += 3 * WD * sizeof(_Float16);
  _Float16* Wtv  = (_Float16*)w;  w += 3 * WD * sizeof(_Float16);
  _Float16* Wto  = (_Float16*)w;  w += 1 * WD * sizeof(_Float16);
  _Float16* ao16 = (_Float16*)w;  w += MD * sizeof(_Float16);      // 16 MB
  float* qbuf = (float*)w;        w += MD * sizeof(float);         // 32 MB
  float* kbuf = (float*)w;        w += MD * sizeof(float);
  float* vbuf = (float*)w;        w += MD * sizeof(float);
  float* gq   = (float*)w;        w += (size_t)MM * 3 * sizeof(float);
  float* gk   = (float*)w;        w += (size_t)MM * 3 * sizeof(float);
  float* gv   = (float*)w;

  // 1) PID scan -> stacked f16 operand [x | integral | derivative]
  pid_scan<<<(BB * DD + 255) / 256, 256, 0, stream>>>(x, Acat);

  // 2) gates (one wave per token)
  dim3 gGrid((MM * 32 + 255) / 256);
  pid_gates<<<gGrid, 256, 0, stream>>>(x, Wg_q, bg_q, gq);
  pid_gates<<<gGrid, 256, 0, stream>>>(x, Wg_k, bg_k, gk);
  pid_gates<<<gGrid, 256, 0, stream>>>(x, Wg_v, bg_v, gv);

  // 3) transpose-convert the 10 weight matrices to f16 [N][K], stacked per proj
  dim3 tGrid(DD / 32, DD / 32), tBlk(32, 8);
  wt_convert<<<tGrid, tBlk, 0, stream>>>(Wp_q, Wtq + 0 * WD, DD, DD);
  wt_convert<<<tGrid, tBlk, 0, stream>>>(Wi_q, Wtq + 1 * WD, DD, DD);
  wt_convert<<<tGrid, tBlk, 0, stream>>>(Wd_q, Wtq + 2 * WD, DD, DD);
  wt_convert<<<tGrid, tBlk, 0, stream>>>(Wp_k, Wtk + 0 * WD, DD, DD);
  wt_convert<<<tGrid, tBlk, 0, stream>>>(Wi_k, Wtk + 1 * WD, DD, DD);
  wt_convert<<<tGrid, tBlk, 0, stream>>>(Wd_k, Wtk + 2 * WD, DD, DD);
  wt_convert<<<tGrid, tBlk, 0, stream>>>(Wp_v, Wtv + 0 * WD, DD, DD);
  wt_convert<<<tGrid, tBlk, 0, stream>>>(Wi_v, Wtv + 1 * WD, DD, DD);
  wt_convert<<<tGrid, tBlk, 0, stream>>>(Wd_v, Wtv + 2 * WD, DD, DD);
  wt_convert<<<tGrid, tBlk, 0, stream>>>(Wo,   Wto,          DD, DD);

  // 4) fused gated projections: q/k/v = g0*x@Wp + g1*integ@Wi + g2*deriv@Wd
  dim3 grid(DD / BN, MM / BM);   // (8, 64)
  gemm_f16<3, true><<<grid, 256, 0, stream>>>(Acat, Wtq, gq, qbuf, MM, DD, DD);
  gemm_f16<3, true><<<grid, 256, 0, stream>>>(Acat, Wtk, gk, kbuf, MM, DD, DD);
  gemm_f16<3, true><<<grid, 256, 0, stream>>>(Acat, Wtv, gv, vbuf, MM, DD, DD);

  // 5) dilated sparse attention (one wave per (b,h,t)) -> f16 rows
  dim3 aGrid(((size_t)BB * HH * TT * 32 + 255) / 256);
  sparse_attn<<<aGrid, 256, 0, stream>>>(qbuf, kbuf, vbuf, positions, mask,
                                         ao16);

  // 6) output projection: out = attn_out @ Wo
  gemm_f16<1, false><<<grid, 256, 0, stream>>>(ao16, Wto, nullptr, outp,
                                               MM, DD, DD);
}
